// LDE_41721312313511
// MI455X (gfx1250) — matrix operand order repaired and checked by reading, not compiled, not run on hardware
//
#include <hip/hip_runtime.h>
#include <stdint.h>

// ---------- types (POD ext-vectors so unions stay trivially constructible) ----
typedef _Float16 h16v __attribute__((ext_vector_type(16)));
typedef float    f8v  __attribute__((ext_vector_type(8)));
typedef int      i4v  __attribute__((ext_vector_type(4)));

union V16 { h16v v; i4v q[2]; unsigned u[8]; _Float16 h[16]; };
union HP  { _Float16 h[2]; unsigned u; };

static __device__ __forceinline__ f8v wmma_f16(h16v a, h16v b, f8v c) {
  // D = A(16x32 f16) * B(32x16 f16) + C(16x16 f32)  -> v_wmma_f32_16x16x32_f16
  return __builtin_amdgcn_wmma_f32_16x16x32_f16(false, a, false, b, (short)0, c,
                                                false, false);
}

#define NPIX 9216   // 96*96
#define NTIL 576    // NPIX/16

// ---------- stage 0a: inputs -> pixel-major f16 [mod][b][px][64] --------------
__global__ void k_pack_x(const float* __restrict__ rgb,
                         const float* __restrict__ dep,
                         _Float16* __restrict__ xT) {
  int idx = blockIdx.x * blockDim.x + threadIdx.x;           // 2*2*9216*64
  if (idx >= 2 * 2 * NPIX * 64) return;
  int c  = idx & 63;
  int px = (idx >> 6) % NPIX;
  int r  = (idx >> 6) / NPIX;
  int b  = r & 1;
  int mod = r >> 1;
  const float* src = mod ? dep : rgb;
  xT[idx] = (_Float16)src[((size_t)b * 64 + c) * NPIX + px];
}

// ---------- stage 0b: weights -> f16 GEMM layouts -----------------------------
__global__ void k_pack_w(const float* __restrict__ w1,  const float* __restrict__ w2,
                         const float* __restrict__ wc1, const float* __restrict__ wc2,
                         const float* __restrict__ wc3, const float* __restrict__ wc4,
                         _Float16* __restrict__ w1p, _Float16* __restrict__ w2p,
                         _Float16* __restrict__ wchp, _Float16* __restrict__ wch4p) {
  int stride = gridDim.x * blockDim.x;
  int t0 = blockIdx.x * blockDim.x + threadIdx.x;
  // w1p[tap][co][ci] from w1[co][ci][3][3]
  for (int i = t0; i < 9 * 64 * 64; i += stride) {
    int ci = i & 63, co = (i >> 6) & 63, tap = i >> 12;
    w1p[i] = (_Float16)w1[(co * 64 + ci) * 9 + tap];
  }
  for (int i = t0; i < 64 * 64; i += stride) w2p[i] = (_Float16)w2[i];
  for (int i = t0; i < 32 * 64; i += stride) {
    wchp[i]        = (_Float16)wc1[i];
    wchp[2048 + i] = (_Float16)wc2[i];
    wchp[4096 + i] = (_Float16)wc3[i];
  }
  for (int i = t0; i < 64 * 32; i += stride) wch4p[i] = (_Float16)wc4[i];
}

// ---------- stage 1: fused conv3x3+PReLU -> conv1x1+PReLU (implicit GEMM) -----
// block = 4 waves; wave w owns output-channel tile w (16 ch), all share a 16-px tile.
__global__ void __launch_bounds__(128) k_trunk(
    const _Float16* __restrict__ xT, const _Float16* __restrict__ w1p,
    const _Float16* __restrict__ w2p,
    const float* __restrict__ b1, const float* __restrict__ b2,
    const float* __restrict__ a1p, const float* __restrict__ a2p,
    _Float16* __restrict__ tHT, float* __restrict__ d0f) {
  __shared__ __align__(16) _Float16 lds[16 * 64];   // [px][co] f16
  int blk  = blockIdx.x;
  int tile = blk % NTIL;
  int b    = (blk / NTIL) & 1;
  int mod  = blk / (NTIL * 2);
  int wid  = threadIdx.x >> 5;
  int lane = threadIdx.x & 31;
  int col  = lane & 15, hi = lane >> 4, k0 = hi * 8;
  int y = tile / 6, xb = (tile % 6) * 16;
  const _Float16* xs = xT + (size_t)(mod * 2 + b) * NPIX * 64;
  float a1 = a1p[0], a2 = a2p[0];

  // conv1 (3x3): K = 64ci * 9taps, chunks of 32
  f8v acc = {0.f, 0.f, 0.f, 0.f, 0.f, 0.f, 0.f, 0.f};
  for (int tap = 0; tap < 9; ++tap) {
    int dy = tap / 3 - 1, dx = tap % 3 - 1;
    int yy = y + dy, xx = xb + col + dx;
    bool ok = (yy >= 0) & (yy < 96) & (xx >= 0) & (xx < 96);
    long poff = (long)(yy * 96 + xx) * 64;
    for (int ch = 0; ch < 2; ++ch) {
      V16 A, Bv;
      const _Float16* ap = w1p + ((size_t)tap * 64 + wid * 16 + col) * 64 + ch * 32;
      A.q[0] = *(const i4v*)(ap + k0);
      A.q[1] = *(const i4v*)(ap + k0 + 16);
      if (ok) {
        const _Float16* bp = xs + poff + ch * 32 + hi * 8;
        Bv.q[0] = *(const i4v*)bp;
        Bv.q[1] = *(const i4v*)(bp + 16);
      } else {
        Bv.q[0] = (i4v){0, 0, 0, 0};
        Bv.q[1] = (i4v){0, 0, 0, 0};
      }
      acc = wmma_f16(A.v, Bv.v, acc);
    }
  }
  // bias + PReLU, stash f16 into LDS [px][co]
  int co0 = wid * 16 + hi * 8;
  V16 p;
  for (int r = 0; r < 8; ++r) {
    float v = acc[r] + b1[co0 + r];
    v = v >= 0.f ? v : a1 * v;
    p.h[r] = (_Float16)v;
  }
  *(i4v*)(&lds[col * 64 + co0]) = p.q[0];
  __syncthreads();

  // conv2 (1x1): K = 64, 2 chunks; B read back transposed from LDS
  f8v acc2 = {0.f, 0.f, 0.f, 0.f, 0.f, 0.f, 0.f, 0.f};
  for (int ch = 0; ch < 2; ++ch) {
    V16 A, Bv;
    const _Float16* ap = w2p + (size_t)(wid * 16 + col) * 64 + ch * 32;
    A.q[0] = *(const i4v*)(ap + k0);
    A.q[1] = *(const i4v*)(ap + k0 + 16);
    const _Float16* lp = &lds[col * 64 + ch * 32 + hi * 8];
    Bv.q[0] = *(const i4v*)lp;
    Bv.q[1] = *(const i4v*)(lp + 16);
    acc2 = wmma_f16(A.v, Bv.v, acc2);
  }
  int pxg = y * 96 + xb + col;
  float vals[8];
  V16 o;
  for (int r = 0; r < 8; ++r) {
    float v = acc2[r] + b2[co0 + r];
    v = v >= 0.f ? v : a2 * v;
    vals[r] = v;
    o.h[r] = (_Float16)v;
  }
  _Float16* outT = tHT + (size_t)(mod * 2 + b) * NPIX * 64 + (size_t)pxg * 64 + co0;
  *(i4v*)outT = o.q[0];
  if (mod == 1) {  // depth trunk kept in f32 for the residual add
    float* dp = d0f + ((size_t)b * 64 + co0) * NPIX + pxg;
    for (int r = 0; r < 8; ++r) dp[(size_t)r * NPIX] = vals[r];
  }
}

// ---------- stage 2: Q/K/V 1x1 convs (M=32, K=64) -----------------------------
// blk -> (b, tile, o); o=0: V=wch1*c0 -> cV[32,n]; o=1: Q -> d1T[n,32]; o=2: K -> d2T[n,32]
__global__ void __launch_bounds__(64) k_qkv(
    const _Float16* __restrict__ tHT, const _Float16* __restrict__ wchp,
    _Float16* __restrict__ d1T, _Float16* __restrict__ d2T,
    _Float16* __restrict__ cV) {
  int blk  = blockIdx.x;
  int o    = blk % 3;
  int tile = (blk / 3) % NTIL;
  int b    = blk / (3 * NTIL);
  int wid  = threadIdx.x >> 5, lane = threadIdx.x & 31;
  int col = lane & 15, hi = lane >> 4, k0 = hi * 8;
  int pxg = tile * 16 + col;
  const _Float16* srcT = tHT + (size_t)((o == 0 ? 0 : 2) + b) * NPIX * 64;
  f8v acc = {0.f, 0.f, 0.f, 0.f, 0.f, 0.f, 0.f, 0.f};
  for (int ch = 0; ch < 2; ++ch) {
    V16 A, Bv;
    const _Float16* ap = wchp + o * 2048 + (size_t)(wid * 16 + col) * 64 + ch * 32;
    A.q[0] = *(const i4v*)(ap + k0);
    A.q[1] = *(const i4v*)(ap + k0 + 16);
    const _Float16* bp = srcT + (size_t)pxg * 64 + ch * 32 + hi * 8;
    Bv.q[0] = *(const i4v*)bp;
    Bv.q[1] = *(const i4v*)(bp + 16);
    acc = wmma_f16(A.v, Bv.v, acc);
  }
  int c0r = wid * 16 + hi * 8;
  if (o == 0) {
    _Float16* vp = cV + (size_t)b * 32 * NPIX + (size_t)c0r * NPIX + pxg;
    for (int r = 0; r < 8; ++r) vp[(size_t)r * NPIX] = (_Float16)acc[r];
  } else {
    _Float16* dp = (o == 1 ? d1T : d2T) + (size_t)b * NPIX * 32 + (size_t)pxg * 32 + c0r;
    V16 pk;
    for (int r = 0; r < 8; ++r) pk.h[r] = (_Float16)acc[r];
    *(i4v*)dp = pk.q[0];
  }
}

// ---------- stage 3: streaming-softmax attention ------------------------------
// One wave per 16-row Q tile. Per 32-col block: 2 S-WMMAs (K=32 head dim),
// pair-lane softmax, 2 accumulating P*V WMMAs. Never materializes [n,n].
__global__ void __launch_bounds__(128) k_attn(
    const _Float16* __restrict__ d1T, const _Float16* __restrict__ d2T,
    const _Float16* __restrict__ cV, _Float16* __restrict__ guidedT) {
  int wid = threadIdx.x >> 5, lane = threadIdx.x & 31;
  int gwave = blockIdx.x * 4 + wid;
  int b = gwave / NTIL, it = gwave % NTIL;
  int ibase = it * 16;
  int col = lane & 15, hi = lane >> 4, k0 = hi * 8;
  const _Float16* q  = d1T + (size_t)b * NPIX * 32;
  const _Float16* kk = d2T + (size_t)b * NPIX * 32;
  const _Float16* vv = cV + (size_t)b * 32 * NPIX;

  V16 Bq;  // Q^T as B operand, resident for the whole sweep
  {
    const _Float16* qp = q + (size_t)(ibase + col) * 32 + hi * 8;
    Bq.q[0] = *(const i4v*)qp;
    Bq.q[1] = *(const i4v*)(qp + 16);
  }
  f8v O0 = {0.f, 0.f, 0.f, 0.f, 0.f, 0.f, 0.f, 0.f};
  f8v O1 = {0.f, 0.f, 0.f, 0.f, 0.f, 0.f, 0.f, 0.f};
  float m = -3.0e38f, l = 0.f;

  for (int jb = 0; jb < NPIX; jb += 32) {
    V16 A1, A2;
    {
      const _Float16* p1 = kk + (size_t)(jb + col) * 32 + k0;
      A1.q[0] = *(const i4v*)p1;
      A1.q[1] = *(const i4v*)(p1 + 16);
      __builtin_prefetch((const void*)(p1 + 32 * 32), 0, 3);  // next K block
    }
    {
      const _Float16* p2 = kk + (size_t)(jb + 16 + col) * 32 + k0;
      A2.q[0] = *(const i4v*)p2;
      A2.q[1] = *(const i4v*)(p2 + 16);
    }
    f8v Z = {0.f, 0.f, 0.f, 0.f, 0.f, 0.f, 0.f, 0.f};
    f8v S0 = wmma_f16(A1.v, Bq.v, Z);   // S^T rows j=jb..jb+15
    f8v S1 = wmma_f16(A2.v, Bq.v, Z);   // S^T rows j=jb+16..jb+31

    // row-i softmax stats live on lane pair (i, i^16)
    float mb = S0[0];
    for (int r = 1; r < 8; ++r) mb = fmaxf(mb, S0[r]);
    for (int r = 0; r < 8; ++r) mb = fmaxf(mb, S1[r]);
    mb = fmaxf(mb, __shfl_xor(mb, 16, 32));
    float mn   = fmaxf(m, mb);
    float corr = __expf(m - mn);
    float e0[8], e1[8], ts = 0.f;
    for (int r = 0; r < 8; ++r) {
      e0[r] = __expf(S0[r] - mn);
      e1[r] = __expf(S1[r] - mn);
      ts += e0[r] + e1[r];
    }
    ts += __shfl_xor(ts, 16, 32);
    l = l * corr + ts;
    m = mn;
    for (int r = 0; r < 8; ++r) { O0[r] *= corr; O1[r] *= corr; }

    // rebuild P^T in B-operand layout: own half + partner half via lane^16 shuffles
    unsigned q0[4], q1[4];
    for (int t = 0; t < 4; ++t) {
      HP h;
      h.h[0] = (_Float16)e0[2 * t]; h.h[1] = (_Float16)e0[2 * t + 1]; q0[t] = h.u;
      h.h[0] = (_Float16)e1[2 * t]; h.h[1] = (_Float16)e1[2 * t + 1]; q1[t] = h.u;
    }
    V16 Bp;
    for (int t = 0; t < 4; ++t) {
      unsigned g0 = __shfl_xor(q0[t], 16, 32);
      unsigned g1 = __shfl_xor(q1[t], 16, 32);
      Bp.u[t]     = hi ? g1 : q0[t];
      Bp.u[4 + t] = hi ? q1[t] : g0;
    }
    // P*V: A = V^T tiles [16c x 32j] from cV[32,n]
    V16 Av0, Av1;
    {
      const _Float16* p = vv + (size_t)col * NPIX + jb + hi * 16;
      Av0.q[0] = *(const i4v*)p;
      Av0.q[1] = *(const i4v*)(p + 8);
    }
    {
      const _Float16* p = vv + (size_t)(16 + col) * NPIX + jb + hi * 16;
      Av1.q[0] = *(const i4v*)p;
      Av1.q[1] = *(const i4v*)(p + 8);
    }
    O0 = wmma_f16(Av0.v, Bp.v, O0);   // O^T rows c=0..15
    O1 = wmma_f16(Av1.v, Bp.v, O1);   // O^T rows c=16..31
  }

  float rl = 1.f / l;
  V16 o0, o1;
  for (int r = 0; r < 8; ++r) {
    o0.h[r] = (_Float16)(O0[r] * rl);
    o1.h[r] = (_Float16)(O1[r] * rl);
  }
  _Float16* gp = guidedT + (size_t)b * NPIX * 32 + (size_t)(ibase + col) * 32 + hi * 8;
  *(i4v*)gp        = o0.q[0];
  *(i4v*)(gp + 16) = o1.q[0];
}

// ---------- stage 4: out = wch4 * guided + d0 (M=64, K=32, one WMMA/tile) -----
__global__ void __launch_bounds__(128) k_final(
    const _Float16* __restrict__ guidedT, const _Float16* __restrict__ wch4p,
    const float* __restrict__ d0f, float* __restrict__ out) {
  int blk  = blockIdx.x;
  int tile = blk % NTIL, b = blk / NTIL;
  int wid = threadIdx.x >> 5, lane = threadIdx.x & 31;
  int col = lane & 15, hi = lane >> 4, k0 = hi * 8;
  int pxg = tile * 16 + col;
  V16 A, Bv;
  const _Float16* ap = wch4p + (size_t)(wid * 16 + col) * 32;
  A.q[0] = *(const i4v*)(ap + k0);
  A.q[1] = *(const i4v*)(ap + k0 + 16);
  const _Float16* bp = guidedT + (size_t)b * NPIX * 32 + (size_t)pxg * 32 + hi * 8;
  Bv.q[0] = *(const i4v*)bp;
  Bv.q[1] = *(const i4v*)(bp + 16);
  f8v acc = {0.f, 0.f, 0.f, 0.f, 0.f, 0.f, 0.f, 0.f};
  acc = wmma_f16(A.v, Bv.v, acc);
  int co0 = wid * 16 + hi * 8;
  const float* dp = d0f + ((size_t)b * 64 + co0) * NPIX + pxg;
  float*       op = out + ((size_t)b * 64 + co0) * NPIX + pxg;
  for (int r = 0; r < 8; ++r)
    op[(size_t)r * NPIX] = acc[r] + dp[(size_t)r * NPIX];
}

// ---------- launch ------------------------------------------------------------
extern "C" void kernel_launch(void* const* d_in, const int* in_sizes, int n_in,
                              void* d_out, int out_size, void* d_ws, size_t ws_size,
                              hipStream_t stream) {
  const float* rgb = (const float*)d_in[0];
  const float* dep = (const float*)d_in[1];
  const float* w1  = (const float*)d_in[2];
  const float* b1  = (const float*)d_in[3];
  const float* a1  = (const float*)d_in[4];
  const float* w2  = (const float*)d_in[5];
  const float* b2  = (const float*)d_in[6];
  const float* a2  = (const float*)d_in[7];
  const float* wc1 = (const float*)d_in[8];
  const float* wc2 = (const float*)d_in[9];
  const float* wc3 = (const float*)d_in[10];
  const float* wc4 = (const float*)d_in[11];

  char* ws = (char*)d_ws;
  const size_t O_XT   = 0;                       // 2*2*9216*64 f16 = 4,718,592 B
  const size_t O_W1P  = O_XT + 4718592;          // 9*64*64 f16
  const size_t O_W2P  = O_W1P + 73728;           // 64*64 f16
  const size_t O_WCHP = O_W2P + 8192;            // 3*32*64 f16
  const size_t O_WC4P = O_WCHP + 12288;          // 64*32 f16
  const size_t O_THT  = O_WC4P + 4096;           // 2*2*9216*64 f16
  const size_t O_D0F  = O_THT + 4718592;         // 2*64*9216 f32
  const size_t O_D1T  = O_D0F + 4718592;         // 2*9216*32 f16
  const size_t O_D2T  = O_D1T + 1179648;
  const size_t O_CV   = O_D2T + 1179648;
  const size_t O_GT   = O_CV + 1179648;

  _Float16* xT    = (_Float16*)(ws + O_XT);
  _Float16* w1p   = (_Float16*)(ws + O_W1P);
  _Float16* w2p   = (_Float16*)(ws + O_W2P);
  _Float16* wchp  = (_Float16*)(ws + O_WCHP);
  _Float16* wch4p = (_Float16*)(ws + O_WC4P);
  _Float16* tHT   = (_Float16*)(ws + O_THT);
  float*    d0f   = (float*)(ws + O_D0F);
  _Float16* d1T   = (_Float16*)(ws + O_D1T);
  _Float16* d2T   = (_Float16*)(ws + O_D2T);
  _Float16* cV    = (_Float16*)(ws + O_CV);
  _Float16* gT    = (_Float16*)(ws + O_GT);

  k_pack_x<<<(2 * 2 * NPIX * 64 + 255) / 256, 256, 0, stream>>>(rgb, dep, xT);
  k_pack_w<<<64, 256, 0, stream>>>(w1, w2, wc1, wc2, wc3, wc4, w1p, w2p, wchp, wch4p);
  k_trunk<<<2 * 2 * NTIL, 128, 0, stream>>>(xT, w1p, w2p, b1, b2, a1, a2, tHT, d0f);
  k_qkv<<<2 * NTIL * 3, 64, 0, stream>>>(tHT, wchp, d1T, d2T, cV);
  k_attn<<<(2 * NTIL) / 4, 128, 0, stream>>>(d1T, d2T, cV, gT);
  k_final<<<2 * NTIL, 128, 0, stream>>>(gT, wch4p, d0f, (float*)d_out);
}
